// RNN2D_36197984370914
// MI455X (gfx1250) — matrix operand order repaired and checked by reading, not compiled, not run on hardware
//
#include <hip/hip_runtime.h>
#include <hip/hip_bf16.h>
#include <math.h>

#define LSIDE 64
#define HID   32
#define BATCH 1024

typedef __attribute__((ext_vector_type(16))) __bf16 v16bf;
typedef __attribute__((ext_vector_type(8)))  float  v8f;
typedef __attribute__((ext_vector_type(4)))  unsigned int u32x4;
typedef __attribute__((ext_vector_type(8)))  int  i32x8;
typedef __attribute__((ext_vector_type(4)))  int  i32x4;

#if defined(__has_builtin)
#if __has_builtin(__builtin_amdgcn_tensor_load_to_lds)
#define HAVE_TDM 1
#endif
#if __has_builtin(__builtin_amdgcn_s_wait_tensorcnt)
#define WAIT_TENSORCNT() __builtin_amdgcn_s_wait_tensorcnt(0)
#else
#define WAIT_TENSORCNT() asm volatile("s_wait_tensorcnt 0x0" ::: "memory")
#endif
#endif
#ifndef WAIT_TENSORCNT
#define WAIT_TENSORCNT() asm volatile("s_wait_tensorcnt 0x0" ::: "memory")
#endif

union BF16Frag {
    v16bf v;
    unsigned short u[16];
    uint4 q[2];
};

__device__ __forceinline__ unsigned short f2bf(float f) {
    union { float f; unsigned int u; } x; x.f = f;
    unsigned int r = x.u + 0x7FFFu + ((x.u >> 16) & 1u);   // round-to-nearest-even
    return (unsigned short)(r >> 16);
}

// One wave (32 lanes) per 16-sample batch tile.
// A = [16 batch rows x 32 K] bf16, B = weight [32 K x 16 N] bf16 (VGPR-resident), C/D = 16x16 f32.
__global__ __launch_bounds__(32)
void rnn2d_wmma_kernel(const int*   __restrict__ x,     // [B,64,64] int32 spins
                       const float* __restrict__ Win,   // [4,32]
                       const float* __restrict__ WcH,   // [32,32]
                       const float* __restrict__ WcV,   // [32,32]
                       const float* __restrict__ bV,    // [32]
                       const float* __restrict__ Wout,  // [32,2]
                       const float* __restrict__ bout,  // [2]
                       float*       __restrict__ out)   // [B]
{
    const int lane = threadIdx.x;       // 0..31
    const int hi   = lane >> 4;         // lane group
    const int n    = lane & 15;         // N column within 16-wide tile / A-row index
    const int tileId = blockIdx.x;      // 0..63
    const int m0 = tileId * 16;

    // LDS: vertical carry grid lives entirely on-chip (critical-path latency),
    // x rows are triple-buffered for TDM prefetch overlap.
    __shared__ unsigned short tile[16 * 32];                 // 1 KB transpose staging
    __shared__ unsigned short carryLDS[LSIDE * 16 * 32];     // 64 KB: per-column A-fragment bytes
    __shared__ int xrows[3][16 * LSIDE];                     // 12 KB: x row cache

    // ---- Weight B-fragments: resident in VGPRs for all 4096 steps ----
    // B 32x16 bf16 layout: lanes 0-15 hold K=0..15 (elem q = K q), lanes 16-31 hold K=16..31.
    BF16Frag bWcH0, bWcH1, bWcV0, bWcV1, bWin0, bWin1, bWout;
    const int kb = hi * 16;
    #pragma unroll
    for (int q = 0; q < 16; ++q) {
        const int k = kb + q;
        bWcH0.u[q] = f2bf(WcH[k * HID + n]);
        bWcH1.u[q] = f2bf(WcH[k * HID + n + 16]);
        bWcV0.u[q] = f2bf(WcV[k * HID + n]);
        bWcV1.u[q] = f2bf(WcV[k * HID + n + 16]);
        bWin0.u[q] = (hi == 0 && q < 4) ? f2bf(Win[q * HID + n])      : (unsigned short)0;
        bWin1.u[q] = (hi == 0 && q < 4) ? f2bf(Win[q * HID + n + 16]) : (unsigned short)0;
        bWout.u[q] = (n < 2) ? f2bf(Wout[k * 2 + n]) : (unsigned short)0;
    }
    const float bv0 = bV[n];
    const float bv1 = bV[n + 16];
    const float bo0 = bout[0];
    const float bo1 = bout[1];

#ifdef HAVE_TDM
    // ---- TDM row loader: 2D tile, 16 rows (samples, stride 4096 ints) x 64 ints ----
    auto tdm_load_row = [&](int r, int buf) {
        const unsigned long long gaddr =
            (unsigned long long)(const void*)(x + (size_t)m0 * (LSIDE * LSIDE) + (size_t)r * LSIDE);
        const unsigned int laddr = (unsigned int)(unsigned long long)(const void*)&xrows[buf][0];
        u32x4 g0;
        g0[0] = 1u;                                   // count=1, is_restore=0, gather off
        g0[1] = laddr;                                // lds_addr
        g0[2] = (unsigned int)(gaddr & 0xFFFFFFFFull);        // global_addr[31:0]
        g0[3] = (unsigned int)((gaddr >> 32) & 0x1FFFFFFull)  // global_addr[56:32]
                | 0x80000000u;                        // type=2 ("image")
        i32x8 g1;
        g1[0] = 0x20000;                              // data_size=2 (4B), no mask/flags
        g1[1] = (LSIDE & 0xFFFF) << 16;               // tensor_dim0 = 64 (bits 79:48)
        g1[2] = (16 & 0xFFFF) << 16;                  // tensor_dim1 = 16 (bits 111:80)
        g1[3] = (LSIDE & 0xFFFF) << 16;               // tile_dim0 = 64 (bits 127:112)
        g1[4] = 16;                                   // tile_dim1 = 16, tile_dim2 = 0
        g1[5] = LSIDE * LSIDE;                        // tensor_dim0_stride = 4096 (low 32)
        g1[6] = 0;                                    // stride high bits
        g1[7] = 0;
        i32x4 gz4 = {0, 0, 0, 0};
        i32x8 gz8 = {0, 0, 0, 0, 0, 0, 0, 0};
        // clang-23 / therock 6-arg form: (g0, g1, g2, g3, g4, cpol)
        __builtin_amdgcn_tensor_load_to_lds(g0, g1, gz4, gz4, gz8, 0);
    };
#else
    // Fallback: coalesced vector loads into the LDS row cache.
    auto tdm_load_row = [&](int r, int buf) {
        #pragma unroll
        for (int it = 0; it < 8; ++it) {
            const int f  = it * 32 + lane;            // 0..255
            const int m  = f >> 4;                    // sample 0..15
            const int c4 = f & 15;                    // int4 index within row
            const int4 v = *(const int4*)&x[(size_t)(m0 + m) * (LSIDE * LSIDE)
                                            + (size_t)r * LSIDE + c4 * 4];
            *(int4*)&xrows[buf][m * LSIDE + c4 * 4] = v;
        }
    };
#endif

    float accLP[8];
    #pragma unroll
    for (int i = 0; i < 8; ++i) accLP[i] = 0.0f;

    BF16Frag hA;   // horizontal carry, bf16 A-fragment (16x32)
    uint4* const carry4 = (uint4*)carryLDS;

    tdm_load_row(0, 0);                               // prime row 0

    for (int r = 0; r < LSIDE; ++r) {
        WAIT_TENSORCNT();                             // row r resident in LDS
        __syncthreads();
        if (r + 1 < LSIDE) tdm_load_row(r + 1, (r + 1) % 3);   // DMA overlaps 64 serial steps

        const int* __restrict__ xcur  = &xrows[r % 3][0];
        const int* __restrict__ xprev = &xrows[(r + 2) % 3][0];   // == (r-1)%3 for r>0

        #pragma unroll
        for (int q = 0; q < 16; ++q) hA.u[q] = 0;     // hcarry resets each row
        const bool even = ((r & 1) == 0);

        for (int t = 0; t < LSIDE; ++t) {
            const int j  = even ? t : (LSIDE - 1 - t);     // spatial column
            const int jp = even ? (j - 1) : (j + 1);       // previous column in scan order

            // ---- newR one-hot A-fragment (exact in bf16): K0,K1 = sH, K2,K3 = sV.
            BF16Frag rA;
            #pragma unroll
            for (int q = 0; q < 16; ++q) rA.u[q] = 0;
            if (hi == 0) {                                  // K=0..3 live in lanes 0-15
                if (t > 0) {
                    const int sH = xcur[n * LSIDE + jp];
                    rA.u[0] = (sH == 0) ? 0x3F80 : 0;
                    rA.u[1] = (sH == 1) ? 0x3F80 : 0;
                }
                if (r > 0) {
                    const int sV = xprev[n * LSIDE + j];
                    rA.u[2] = (sV == 0) ? 0x3F80 : 0;
                    rA.u[3] = (sV == 1) ? 0x3F80 : 0;
                }
            }

            // ---- vertical carry A-fragment straight from LDS (A-layout bytes)
            const int slot = (j * 32 + lane) * 2;          // uint4 units
            BF16Frag cA;
            if (r > 0) {
                cA.q[0] = carry4[slot];
                cA.q[1] = carry4[slot + 1];
            } else {
                #pragma unroll
                for (int q = 0; q < 16; ++q) cA.u[q] = 0;
            }

            // ---- bV + newR*Win + hcarry*WcH + cV*WcV (f32-accumulated WMMA chain)
            v8f c0, c1;
            #pragma unroll
            for (int i = 0; i < 8; ++i) { c0[i] = bv0; c1[i] = bv1; }
            c0 = __builtin_amdgcn_wmma_f32_16x16x32_bf16(false, rA.v, false, bWin0.v, (short)0, c0, false, false);
            c1 = __builtin_amdgcn_wmma_f32_16x16x32_bf16(false, rA.v, false, bWin1.v, (short)0, c1, false, false);
            c0 = __builtin_amdgcn_wmma_f32_16x16x32_bf16(false, hA.v, false, bWcH0.v, (short)0, c0, false, false);
            c1 = __builtin_amdgcn_wmma_f32_16x16x32_bf16(false, hA.v, false, bWcH1.v, (short)0, c1, false, false);
            c0 = __builtin_amdgcn_wmma_f32_16x16x32_bf16(false, cA.v, false, bWcV0.v, (short)0, c0, false, false);
            c1 = __builtin_amdgcn_wmma_f32_16x16x32_bf16(false, cA.v, false, bWcV1.v, (short)0, c1, false, false);

            // ---- ELU
            #pragma unroll
            for (int i = 0; i < 8; ++i) {
                float a0 = c0[i]; c0[i] = (a0 > 0.0f) ? a0 : (__expf(a0) - 1.0f);
                float a1 = c1[i]; c1[i] = (a1 > 0.0f) ? a1 : (__expf(a1) - 1.0f);
            }

            // ---- C layout -> bf16 row-major LDS tile -> A layout (transpose roundtrip)
            #pragma unroll
            for (int i = 0; i < 8; ++i) {
                const int m = i + 8 * hi;
                tile[m * 32 + n]      = f2bf(c0[i]);
                tile[m * 32 + n + 16] = f2bf(c1[i]);
            }
            __syncthreads();
            {
                const int k0 = hi ? 16 : 0;
                const int k1 = hi ? 24 : 8;
                hA.q[0] = *(const uint4*)&tile[n * 32 + k0];
                hA.q[1] = *(const uint4*)&tile[n * 32 + k1];
            }
            __syncthreads();

            // ---- persist vertical carry (A-fragment order) for row r+1 — stays in LDS
            carry4[slot]     = hA.q[0];
            carry4[slot + 1] = hA.q[1];

            // ---- output head: newC @ Wout; out0/out1 land on lanes n=0/1 of each group
            v8f co;
            #pragma unroll
            for (int i = 0; i < 8; ++i) co[i] = 0.0f;
            co = __builtin_amdgcn_wmma_f32_16x16x32_bf16(false, hA.v, false, bWout.v, (short)0, co, false, false);

            #pragma unroll
            for (int i = 0; i < 8; ++i) {
                float o0 = co[i];
                float o1 = __shfl_xor(o0, 1, 32);          // all lanes participate
                if (n == 0) {                              // lanes 0 (m=i) and 16 (m=8+i)
                    o0 += bo0; o1 += bo1;
                    const int m = i + 8 * hi;
                    const int s = xcur[m * LSIDE + j];
                    const float mx  = fmaxf(o0, o1);
                    const float lse = mx + __logf(__expf(o0 - mx) + __expf(o1 - mx));
                    float lp = (s ? o1 : o0) - lse;
                    lp = (lp != lp) ? -35.0f : lp;         // nan_to_num
                    accLP[i] += lp;
                }
            }
        }
    }

    if (n == 0) {
        #pragma unroll
        for (int i = 0; i < 8; ++i)
            out[m0 + i + 8 * hi] = 0.5f * accLP[i];
    }
}

extern "C" void kernel_launch(void* const* d_in, const int* in_sizes, int n_in,
                              void* d_out, int out_size, void* d_ws, size_t ws_size,
                              hipStream_t stream) {
    const int*   x    = (const int*)  d_in[0];
    const float* Win  = (const float*)d_in[1];
    const float* WcH  = (const float*)d_in[2];
    const float* WcV  = (const float*)d_in[3];
    const float* bVv  = (const float*)d_in[4];
    const float* Wout = (const float*)d_in[5];
    const float* bout = (const float*)d_in[6];
    (void)d_ws; (void)ws_size; (void)in_sizes; (void)n_in; (void)out_size;

    // 64 batch tiles of 16 samples, one wave32 per tile; all recurrent state in LDS.
    rnn2d_wmma_kernel<<<BATCH / 16, 32, 0, stream>>>(
        x, Win, WcH, WcV, bVv, Wout, bout, (float*)d_out);
}
